// EALSTM_23373212024883
// MI455X (gfx1250) — matrix operand order repaired and checked by reading, not compiled, not run on hardware
//
#include <hip/hip_runtime.h>
#include <math.h>

// EA-LSTM fused persistent kernel for gfx1250 (MI455X), round 2.
// Fix vs round 1: weight B-fragments split between VGPRs (K frags 0..4,
// 120 VGPRs/wave) and LDS (K frags 5..9, 240 KB pre-swizzled into WMMA
// fragment layout) to stay under the 256-VGPR cliff and kill the
// scratch spills/reloads that round 1 put on the serial T critical path.
// Uses CDNA5's 320 KB/WGP LDS (dynamic shared allocation).

typedef __attribute__((ext_vector_type(16))) __bf16 v16bf;
typedef __attribute__((ext_vector_type(8)))  float  v8f;
typedef __attribute__((ext_vector_type(4)))  unsigned int v4u;

#define B_  1024
#define T_  365
#define FD  64
#define H_  256
#define G3  768
#define KT  320   // H + FD : concatenated K for [h | x_t]
#define D1_ 128

#define NREG_K 5                      // K-fragments kept in VGPRs (k = 0..4)
#define NLDS_K 5                      // K-fragments kept in LDS   (k = 5..9)
#define A_ELEMS (16 * KT)             // 5120 ushorts  (10240 B)
#define W_ELEMS (48 * NLDS_K * 512)   // 122880 ushorts (245760 B) swizzled frags
#define SMEM_BYTES ((A_ELEMS + W_ELEMS) * 2 + 16 * D1_ * 4)   // 264192 B

union AF { v16bf v; v4u q[2]; };
union WF { v16bf v; unsigned short u[16]; };

static __device__ __forceinline__ unsigned short f2bf(float f) {
  union { float f; unsigned int u; } a; a.f = f;
  unsigned int u = a.u;
  u += 0x7FFFu + ((u >> 16) & 1u);   // round-to-nearest-even
  return (unsigned short)(u >> 16);
}
static __device__ __forceinline__ float sigf(float x) {
  return 1.0f / (1.0f + __expf(-x));
}

__global__ __launch_bounds__(512, 1)
void ealstm_persistent_kernel(const float* __restrict__ x,        // [B,T,64]
                              const float* __restrict__ latlons,  // [B,2]
                              const float* __restrict__ Wih,      // [64,768]
                              const float* __restrict__ Whh,      // [256,768]
                              const float* __restrict__ Wsh,      // [2,256]
                              const float* __restrict__ bias,     // [768]
                              const float* __restrict__ bias_s,   // [256]
                              const float* __restrict__ W1,       // [256,128]
                              const float* __restrict__ b1,       // [128]
                              const float* __restrict__ W2,       // [128,1]
                              const float* __restrict__ b2,       // [1]
                              float* __restrict__ out)            // [B,1]
{
  extern __shared__ __align__(16) unsigned short smem[];
  unsigned short* a_buf = smem;                 // bf16 [16][KT]   : [h | x_t]
  unsigned short* wbuf  = smem + A_ELEMS;       // swizzled B-frags, k = 5..9
  float* zbuf = (float*)(smem + A_ELEMS + W_ELEMS);   // head activations

  const int tid   = threadIdx.x;        // 0..511 (16 wave32 waves)
  const int lane  = tid & 31;
  const int wave  = tid >> 5;           // 0..15
  const int half  = lane >> 4;          // lane half (ISA fragment layouts)
  const int n     = lane & 15;
  const int wcol  = wave << 4;          // hidden column base owned by this wave
  const int col16 = wcol + n;           // 0..255
  const int row0  = blockIdx.x << 4;    // batch tile base
  const int mrow  = half * 8;           // C/D row base for this lane half

  // ---- Preload weight B-fragments: k=0..4 -> VGPRs, k=5..9 -> swizzled LDS --
  // B-frag layout: element j = W[kbase + half*16 + j][ncol] (16 consecutive K rows)
  v16bf wfrag[3][NREG_K];
  #pragma unroll
  for (int tl = 0; tl < 3; ++tl) {
    const int ncol = tl * H_ + col16;   // gate tl, this wave's columns
    #pragma unroll
    for (int k = 0; k < 10; ++k) {
      WF wf;
      #pragma unroll
      for (int j = 0; j < 16; ++j) {
        const int kk = k * 32 + half * 16 + j;   // concatenated K index
        const float w = (kk < H_) ? Whh[(size_t)kk * G3 + ncol]
                                  : Wih[(size_t)(kk - H_) * G3 + ncol];
        wf.u[j] = f2bf(w);
      }
      if (k < NREG_K) {
        wfrag[tl][k] = wf.v;
      } else {
        // swizzled store: per-(tile,k) fragment is lane-contiguous 32 B
        unsigned short* wp =
            &wbuf[(((wave * 3 + tl) * NLDS_K) + (k - NREG_K)) * 512 + lane * 16];
        #pragma unroll
        for (int j = 0; j < 16; ++j) wp[j] = wf.u[j];
      }
    }
  }
  const float bv0 = bias[0 * H_ + col16];   // f gate bias
  const float bv1 = bias[1 * H_ + col16];   // o gate bias
  const float bv2 = bias[2 * H_ + col16];   // g gate bias

  // ---- Static input gate (computed once), c state, h state: all in VGPRs ----
  float ig[8], cf[8], hnew[8];
  {
    const float ws0 = Wsh[col16];
    const float ws1 = Wsh[H_ + col16];
    const float bs  = bias_s[col16];
    #pragma unroll
    for (int r = 0; r < 8; ++r) {
      const int m = mrow + r;
      const float l0 = latlons[(size_t)(row0 + m) * 2 + 0];
      const float l1 = latlons[(size_t)(row0 + m) * 2 + 1];
      ig[r]   = sigf(l0 * ws0 + l1 * ws1 + bs);
      cf[r]   = 0.f;
      hnew[r] = 0.f;   // h_0 = 0 (written into LDS at first iteration)
    }
  }

  const int xm = tid >> 6;   // staging coords for the prefetch address
  const int xf = tid & 63;

  // ---------------------------- Recurrence over T ----------------------------
  for (int t = 0; t < T_; ++t) {
    // Stage x_t (f32 -> bf16) into a_buf[:, 256:320]; 1024 elems / 512 threads.
    {
      int e = tid;
      #pragma unroll
      for (int it = 0; it < 2; ++it) {
        const int m = e >> 6, f = e & 63;
        const float xv = x[((size_t)(row0 + m) * T_ + t) * FD + f];
        a_buf[m * KT + H_ + f] = f2bf(xv);
        e += 512;
      }
    }
    // Stage h_t (from previous iteration's registers) into a_buf[:, 0:256].
    #pragma unroll
    for (int r = 0; r < 8; ++r)
      a_buf[(mrow + r) * KT + col16] = f2bf(hnew[r]);

    if (t + 1 < T_)  // hide next step's x fetch latency (global_prefetch_b8)
      __builtin_prefetch(&x[((size_t)(row0 + xm) * T_ + (t + 1)) * FD + xf], 0, 1);

    __syncthreads();   // (also covers the one-time wbuf preamble writes at t==0)

    // gates[16 x 768] tile slice = [h | x_t] (16x320) @ [[Whh],[Wih]] (320x768)
    v8f acc0 = {}, acc1 = {}, acc2 = {};

    // k = 0..4 : B-fragments resident in VGPRs
    #pragma unroll
    for (int k = 0; k < NREG_K; ++k) {
      AF af;
      const unsigned short* bp = &a_buf[n * KT + k * 32 + half * 16];
      af.q[0] = *(const v4u*)(bp);       // ds_load_b128
      af.q[1] = *(const v4u*)(bp + 8);   // ds_load_b128
      acc0 = __builtin_amdgcn_wmma_f32_16x16x32_bf16(false, af.v, false, wfrag[0][k], (short)0, acc0, false, false);
      acc1 = __builtin_amdgcn_wmma_f32_16x16x32_bf16(false, af.v, false, wfrag[1][k], (short)0, acc1, false, false);
      acc2 = __builtin_amdgcn_wmma_f32_16x16x32_bf16(false, af.v, false, wfrag[2][k], (short)0, acc2, false, false);
    }
    // k = 5..9 : B-fragments streamed from swizzled LDS (2x ds_load_b128 each)
    #pragma unroll
    for (int k = 0; k < NLDS_K; ++k) {
      AF af;
      const unsigned short* bp = &a_buf[n * KT + (NREG_K + k) * 32 + half * 16];
      af.q[0] = *(const v4u*)(bp);
      af.q[1] = *(const v4u*)(bp + 8);
      AF bf0, bf1, bf2;
      const unsigned short* w0 = &wbuf[(((wave * 3 + 0) * NLDS_K) + k) * 512 + lane * 16];
      const unsigned short* w1 = &wbuf[(((wave * 3 + 1) * NLDS_K) + k) * 512 + lane * 16];
      const unsigned short* w2 = &wbuf[(((wave * 3 + 2) * NLDS_K) + k) * 512 + lane * 16];
      bf0.q[0] = *(const v4u*)(w0); bf0.q[1] = *(const v4u*)(w0 + 8);
      bf1.q[0] = *(const v4u*)(w1); bf1.q[1] = *(const v4u*)(w1 + 8);
      bf2.q[0] = *(const v4u*)(w2); bf2.q[1] = *(const v4u*)(w2 + 8);
      acc0 = __builtin_amdgcn_wmma_f32_16x16x32_bf16(false, af.v, false, bf0.v, (short)0, acc0, false, false);
      acc1 = __builtin_amdgcn_wmma_f32_16x16x32_bf16(false, af.v, false, bf1.v, (short)0, acc1, false, false);
      acc2 = __builtin_amdgcn_wmma_f32_16x16x32_bf16(false, af.v, false, bf2.v, (short)0, acc2, false, false);
    }

    // Wave-local elementwise LSTM update (f, o, g all live on this wave's cols)
    #pragma unroll
    for (int r = 0; r < 8; ++r) {
      const float fg = sigf(acc0[r] + bv0);
      const float og = sigf(acc1[r] + bv1);
      const float gg = tanhf(acc2[r] + bv2);
      cf[r]   = fg * cf[r] + ig[r] * gg;
      hnew[r] = og * tanhf(cf[r]);
    }
    __syncthreads();   // all A-frag reads done before h region is rewritten
  }

  // ---- Final h -> LDS (bf16) for the dense head GEMM ----
  #pragma unroll
  for (int r = 0; r < 8; ++r)
    a_buf[(mrow + r) * KT + col16] = f2bf(hnew[r]);
  __syncthreads();

  // z = relu(h_last @ W1 + b1): waves 0..7 own 16 of the 128 output columns.
  if (wave < 8) {   // wave-uniform branch; EXEC stays all-ones inside
    const int ncolh = wcol + n;  // 0..127
    v8f accz = {};
    #pragma unroll
    for (int k = 0; k < 8; ++k) {
      AF af;
      const unsigned short* bp = &a_buf[n * KT + k * 32 + half * 16];
      af.q[0] = *(const v4u*)(bp);
      af.q[1] = *(const v4u*)(bp + 8);
      WF wf;
      #pragma unroll
      for (int j = 0; j < 16; ++j) {
        const int kk = k * 32 + half * 16 + j;
        wf.u[j] = f2bf(W1[(size_t)kk * D1_ + ncolh]);
      }
      accz = __builtin_amdgcn_wmma_f32_16x16x32_bf16(false, af.v, false, wf.v, (short)0, accz, false, false);
    }
    const float b1v = b1[ncolh];
    #pragma unroll
    for (int r = 0; r < 8; ++r) {
      const float z = accz[r] + b1v;
      zbuf[(mrow + r) * D1_ + ncolh] = z > 0.f ? z : 0.f;
    }
  }
  __syncthreads();

  // out = z @ W2 + b2 : one lane per batch row.
  if (tid < 16) {
    float s = b2[0];
    for (int k = 0; k < D1_; ++k) s += zbuf[tid * D1_ + k] * W2[k];
    out[row0 + tid] = s;
  }
}

extern "C" void kernel_launch(void* const* d_in, const int* in_sizes, int n_in,
                              void* d_out, int out_size, void* d_ws, size_t ws_size,
                              hipStream_t stream) {
  (void)in_sizes; (void)n_in; (void)out_size; (void)d_ws; (void)ws_size;
  const float* x       = (const float*)d_in[0];
  const float* latlons = (const float*)d_in[1];
  const float* Wih     = (const float*)d_in[2];
  const float* Whh     = (const float*)d_in[3];
  const float* Wsh     = (const float*)d_in[4];
  const float* bias    = (const float*)d_in[5];
  const float* bias_s  = (const float*)d_in[6];
  const float* W1      = (const float*)d_in[7];
  const float* b1      = (const float*)d_in[8];
  const float* W2      = (const float*)d_in[9];
  const float* b2      = (const float*)d_in[10];
  float* out = (float*)d_out;

  // Allow the large dynamic-LDS allocation (CDNA5: 320 KB per WGP/workgroup).
  static bool attr_set = false;  // idempotent host-side attribute, not a stream op
  if (!attr_set) {
    (void)hipFuncSetAttribute((const void*)ealstm_persistent_kernel,
                              hipFuncAttributeMaxDynamicSharedMemorySize,
                              SMEM_BYTES);
    attr_set = true;
  }

  dim3 grid(B_ / 16);   // 64 workgroups, each owns 16 batch rows
  dim3 block(512);      // 16 wave32 waves
  ealstm_persistent_kernel<<<grid, block, SMEM_BYTES, stream>>>(
      x, latlons, Wih, Whh, Wsh, bias, bias_s, W1, b1, W2, b2, out);
}
